// GPT_12558484374093
// MI455X (gfx1250) — matrix operand order repaired and checked by reading, not compiled, not run on hardware
//
#include <hip/hip_runtime.h>

// ---------------- constants ----------------
constexpr int cV  = 50257;
constexpr int cE  = 768;
constexpr int cT  = 1024;
constexpr int cB  = 4;
constexpr int cL  = 12;
constexpr int cH  = 12;
constexpr int cD  = 64;
constexpr int cFF = 3072;
constexpr int cM  = cB * cT;   // 4096 rows

// ---------------- types ----------------
typedef unsigned short u16;
typedef __attribute__((ext_vector_type(8)))  unsigned short u16x8;
typedef __attribute__((ext_vector_type(16))) __bf16         bf16x16;
typedef __attribute__((ext_vector_type(8)))  float          f32x8;

union FragU { u16x8 h[2]; bf16x16 v; };

__device__ __forceinline__ u16 f2bf(float f) {
    unsigned u = __float_as_uint(f);
    u += 0x7FFFu + ((u >> 16) & 1u);   // round-to-nearest-even
    return (u16)(u >> 16);
}

#define WMMA_BF16(A, B, C) \
    __builtin_amdgcn_wmma_f32_16x16x32_bf16(false, (A), false, (B), (short)0, (C), false, false)

// A fragment: 16x32 bf16, M rows striped per lane (ISA 16-bit A layout).
// lanes 0-15: M=lane, K halves {0..7,16..23}; lanes 16-31: M=lane-16, K {8..15,24..31}
__device__ __forceinline__ bf16x16 load_a_frag(const u16* A, long lda, int row0, int k0, int lane) {
    int m  = lane & 15;
    int kb = (lane >> 4) << 3;            // 0 or 8
    const u16* p = A + (long)(row0 + m) * lda + k0 + kb;
    FragU u;
    u.h[0] = *(const u16x8*)(p);          // K = k0+kb .. +7
    u.h[1] = *(const u16x8*)(p + 16);     // K = k0+kb+16 .. +23
    return u.v;
}

// B fragment from BT (B transposed, rows = N, cols = K), 32x16 bf16 B layout.
// lanes 0-15: N=lane, K {0..7,16..23}; lanes 16-31: N=lane-16, K {8..15,24..31}
__device__ __forceinline__ bf16x16 load_bt_frag(const u16* BT, long ldbt, int col0, int k0,
                                                int nmax, int lane) {
    int n = col0 + (lane & 15);
    n = (n < nmax) ? n : (nmax - 1);      // clamp for ragged N (lm_head)
    int kb = (lane >> 4) << 3;            // 0 or 8
    const u16* p = BT + (long)n * ldbt + k0 + kb;
    FragU u;
    u.h[0] = *(const u16x8*)(p);          // K = k0+kb .. +7
    u.h[1] = *(const u16x8*)(p + 16);     // K = k0+kb+16 .. +23
    return u.v;
}

// ---------------- elementwise kernels ----------------
__global__ __launch_bounds__(256) void embed_kernel(const int* __restrict__ idx,
                                                    const float* __restrict__ wte,
                                                    const float* __restrict__ wpe,
                                                    float* __restrict__ x) {
    int id = blockIdx.x * 256 + threadIdx.x;
    if (id >= cM * cE) return;
    int row = id / cE, e = id % cE;
    int t = row & (cT - 1);
    int tok = idx[row];
    x[id] = wte[(size_t)tok * cE + e] + wpe[(size_t)t * cE + e];
}

__global__ __launch_bounds__(256) void f32_to_bf16_kernel(const float* __restrict__ in,
                                                          u16* __restrict__ out, int n) {
    int id = blockIdx.x * 256 + threadIdx.x;
    if (id < n) out[id] = f2bf(in[id]);
}

// W [K,N] f32  ->  out [N,K] bf16 (i.e. B^T, row stride K)
__global__ __launch_bounds__(256) void transp_conv_kernel(const float* __restrict__ W,
                                                          u16* __restrict__ out, int K, int N) {
    int id = blockIdx.x * 256 + threadIdx.x;
    if (id >= K * N) return;
    int k = id / N, n = id % N;
    out[(size_t)n * K + k] = f2bf(W[id]);
}

// vT[b][h][d][t] = qkv[b*1024+t][1536 + h*64 + d]
__global__ __launch_bounds__(256) void transpose_v_kernel(const u16* __restrict__ qkv,
                                                          u16* __restrict__ vT) {
    int id = blockIdx.x * 256 + threadIdx.x;
    if (id >= cB * cH * cD * cT) return;
    int t  = id & (cT - 1);
    int d  = (id >> 10) & (cD - 1);
    int bh = id >> 16;
    int b = bh / cH, h = bh % cH;
    vT[id] = qkv[(size_t)(b * cT + t) * (3 * cE) + 2 * cE + h * cD + d];
}

// ---------------- LayerNorm (f32 in -> bf16 out) ----------------
__global__ __launch_bounds__(256) void ln_kernel(const float* __restrict__ x,
                                                 const float* __restrict__ w,
                                                 const float* __restrict__ b,
                                                 u16* __restrict__ out) {
    __shared__ float red[256];
    int row = blockIdx.x, tid = threadIdx.x;
    const float* xr = x + (size_t)row * cE;
    float v0 = xr[tid], v1 = xr[tid + 256], v2 = xr[tid + 512];
    red[tid] = v0 + v1 + v2;
    __syncthreads();
    for (int o = 128; o > 0; o >>= 1) { if (tid < o) red[tid] += red[tid + o]; __syncthreads(); }
    float mean = red[0] * (1.0f / cE);
    __syncthreads();
    float d0 = v0 - mean, d1 = v1 - mean, d2 = v2 - mean;
    red[tid] = d0 * d0 + d1 * d1 + d2 * d2;
    __syncthreads();
    for (int o = 128; o > 0; o >>= 1) { if (tid < o) red[tid] += red[tid + o]; __syncthreads(); }
    float rs = rsqrtf(red[0] * (1.0f / cE) + 1e-5f);
    u16* orow = out + (size_t)row * cE;
    orow[tid]       = f2bf(d0 * rs * w[tid]       + b[tid]);
    orow[tid + 256] = f2bf(d1 * rs * w[tid + 256] + b[tid + 256]);
    orow[tid + 512] = f2bf(d2 * rs * w[tid + 512] + b[tid + 512]);
}

// ---------------- GEMM: C = A[M,K](bf16) x B (via BT[N,K] bf16) ----------------
// mode: 0 = f32 store, 1 = f32 residual add (Cf += v), 2 = bf16 store, 3 = GELU -> bf16
__global__ __launch_bounds__(256) void gemm_bf16_kernel(const u16* __restrict__ A,
                                                        const u16* __restrict__ BT,
                                                        const float* __restrict__ bias,
                                                        float* __restrict__ Cf,
                                                        u16* __restrict__ Cb,
                                                        int M, int N, int K,
                                                        long ldc, int mode) {
    int lane = threadIdx.x & 31;
    int wave = threadIdx.x >> 5;
    int row0 = blockIdx.y * 256 + wave * 32;
    int col0 = blockIdx.x * 64;
    (void)M;

    f32x8 acc[2][4] = {};
    for (int k0 = 0; k0 < K; k0 += 32) {
        bf16x16 a0 = load_a_frag(A, K, row0,      k0, lane);
        bf16x16 a1 = load_a_frag(A, K, row0 + 16, k0, lane);
#pragma unroll
        for (int j = 0; j < 4; ++j) {
            bf16x16 bfr = load_bt_frag(BT, K, col0 + j * 16, k0, N, lane);
            acc[0][j] = WMMA_BF16(a0, bfr, acc[0][j]);
            acc[1][j] = WMMA_BF16(a1, bfr, acc[1][j]);
        }
    }

    int laneN = lane & 15;
    int mb = (lane >> 4) << 3;
#pragma unroll
    for (int i = 0; i < 2; ++i) {
#pragma unroll
        for (int j = 0; j < 4; ++j) {
            int c = col0 + j * 16 + laneN;
            if (c >= N) continue;
            float bv = bias ? bias[c] : 0.0f;
#pragma unroll
            for (int r = 0; r < 8; ++r) {
                int row = row0 + i * 16 + mb + r;
                float v = acc[i][j][r] + bv;
                if (mode == 0) {
                    Cf[(size_t)row * ldc + c] = v;
                } else if (mode == 1) {
                    Cf[(size_t)row * ldc + c] += v;
                } else if (mode == 2) {
                    Cb[(size_t)row * ldc + c] = f2bf(v);
                } else {
                    float g = 0.5f * v * (1.0f + erff(v * 0.70710678118654752f));
                    Cb[(size_t)row * ldc + c] = f2bf(g);
                }
            }
        }
    }
}

// ---------------- flash attention (WMMA), one wave = 16 query rows ----------------
__global__ __launch_bounds__(128) void attn_kernel(const u16* __restrict__ qkv,
                                                   const u16* __restrict__ vT,
                                                   u16* __restrict__ att) {
    __shared__ __align__(16) u16 ptile[4][16 * 32];
    int lane = threadIdx.x & 31;
    int wave = threadIdx.x >> 5;
    int bid  = blockIdx.x;
    int qblk = bid & 15;
    int bh   = bid >> 4;
    int b = bh / cH, h = bh % cH;
    int qr0 = (qblk * 4 + wave) * 16;

    const u16* qbase = qkv + (size_t)(b * cT) * (3 * cE) + h * cD;
    const u16* kbase = qbase + cE;
    const u16* vtb   = vT + (size_t)bh * cD * cT;

    int laneN = lane & 15;
    int mb = (lane >> 4) << 3;

    bf16x16 aq0 = load_a_frag(qbase, 3 * cE, qr0, 0,  lane);
    bf16x16 aq1 = load_a_frag(qbase, 3 * cE, qr0, 32, lane);

    f32x8 O[4] = {};
    float mrun[8], lrun[8];
#pragma unroll
    for (int r = 0; r < 8; ++r) { mrun[r] = -1e30f; lrun[r] = 0.0f; }

    u16* myp = &ptile[wave][0];

    for (int kb = 0; kb <= qr0 + 15; kb += 32) {
        f32x8 s[2] = {};
#pragma unroll
        for (int n = 0; n < 2; ++n) {
            bf16x16 bk = load_bt_frag(kbase, 3 * cE, kb + n * 16, 0, cT, lane);
            s[n] = WMMA_BF16(aq0, bk, s[n]);
            bk = load_bt_frag(kbase, 3 * cE, kb + n * 16, 32, cT, lane);
            s[n] = WMMA_BF16(aq1, bk, s[n]);
        }
#pragma unroll
        for (int r = 0; r < 8; ++r) {
            int qi = qr0 + mb + r;
            float s0 = s[0][r] * 0.125f;
            float s1 = s[1][r] * 0.125f;
            if (kb + laneN      > qi) s0 = -1e30f;
            if (kb + 16 + laneN > qi) s1 = -1e30f;
            float mx = fmaxf(s0, s1);
            mx = fmaxf(mx, __shfl_xor(mx, 1));
            mx = fmaxf(mx, __shfl_xor(mx, 2));
            mx = fmaxf(mx, __shfl_xor(mx, 4));
            mx = fmaxf(mx, __shfl_xor(mx, 8));
            float mnew  = fmaxf(mrun[r], mx);
            float alpha = __expf(mrun[r] - mnew);
            mrun[r] = mnew;
            float p0 = __expf(s0 - mnew);
            float p1 = __expf(s1 - mnew);
            float ps = p0 + p1;
            ps += __shfl_xor(ps, 1);
            ps += __shfl_xor(ps, 2);
            ps += __shfl_xor(ps, 4);
            ps += __shfl_xor(ps, 8);
            lrun[r] = lrun[r] * alpha + ps;
            O[0][r] *= alpha; O[1][r] *= alpha; O[2][r] *= alpha; O[3][r] *= alpha;
            myp[(mb + r) * 32 + laneN]      = f2bf(p0);
            myp[(mb + r) * 32 + 16 + laneN] = f2bf(p1);
        }
        asm volatile("s_wait_dscnt 0" ::: "memory");   // per-wave LDS store->load fence
        bf16x16 aP = load_a_frag(myp, 32, 0, 0, lane);
#pragma unroll
        for (int j = 0; j < 4; ++j) {
            bf16x16 bv = load_bt_frag(vtb, cT, j * 16, kb, cD, lane);
            O[j] = WMMA_BF16(aP, bv, O[j]);
        }
    }

    u16* obase = att + (size_t)(b * cT + qr0) * cE + h * cD;
#pragma unroll
    for (int j = 0; j < 4; ++j)
#pragma unroll
        for (int r = 0; r < 8; ++r)
            obase[(size_t)(mb + r) * cE + j * 16 + laneN] = f2bf(O[j][r] / lrun[r]);
}

// ---------------- loss ----------------
__global__ __launch_bounds__(256) void loss_row_kernel(const float* __restrict__ logits,
                                                       const int* __restrict__ targets,
                                                       float* __restrict__ nll) {
    __shared__ float red[256];
    int row = blockIdx.x, tid = threadIdx.x;
    const float* lr = logits + (size_t)row * cV;
    float mx = -3.4e38f;
    for (int i = tid; i < cV; i += 256) mx = fmaxf(mx, lr[i]);
    red[tid] = mx;
    __syncthreads();
    for (int o = 128; o > 0; o >>= 1) { if (tid < o) red[tid] = fmaxf(red[tid], red[tid + o]); __syncthreads(); }
    mx = red[0];
    __syncthreads();
    float s = 0.0f;
    for (int i = tid; i < cV; i += 256) s += expf(lr[i] - mx);
    red[tid] = s;
    __syncthreads();
    for (int o = 128; o > 0; o >>= 1) { if (tid < o) red[tid] += red[tid + o]; __syncthreads(); }
    s = red[0];
    if (tid == 0) {
        int tgt = targets[row];
        if (tgt != -1) {
            int t = tgt < 0 ? 0 : tgt;
            nll[row] = -(lr[t] - mx - logf(s));
        } else {
            nll[row] = 0.0f;
        }
    }
}

__global__ __launch_bounds__(256) void loss_reduce_kernel(const float* __restrict__ nll,
                                                          const int* __restrict__ targets,
                                                          float* __restrict__ out_loss) {
    __shared__ float rs[256];
    __shared__ int   rc[256];
    int tid = threadIdx.x;
    float s = 0.0f; int c = 0;
    for (int i = tid; i < cM; i += 256) { s += nll[i]; c += (targets[i] != -1); }
    rs[tid] = s; rc[tid] = c;
    __syncthreads();
    for (int o = 128; o > 0; o >>= 1) {
        if (tid < o) { rs[tid] += rs[tid + o]; rc[tid] += rc[tid + o]; }
        __syncthreads();
    }
    if (tid == 0) out_loss[0] = rs[0] / (float)(rc[0] > 0 ? rc[0] : 1);
}

// ---------------- host launch ----------------
extern "C" void kernel_launch(void* const* d_in, const int* in_sizes, int n_in,
                              void* d_out, int out_size, void* d_ws, size_t ws_size,
                              hipStream_t stream) {
    (void)in_sizes; (void)n_in; (void)out_size; (void)ws_size;

    const int*   idx     = (const int*)d_in[0];
    const int*   targets = (const int*)d_in[1];
    const float* wte     = (const float*)d_in[2];
    const float* wpe     = (const float*)d_in[3];
    const float* ln1_w   = (const float*)d_in[4];
    const float* ln1_b   = (const float*)d_in[5];
    const float* w_qkv   = (const float*)d_in[6];
    const float* b_qkv   = (const float*)d_in[7];
    const float* w_ap    = (const float*)d_in[8];
    const float* b_ap    = (const float*)d_in[9];
    const float* ln2_w   = (const float*)d_in[10];
    const float* ln2_b   = (const float*)d_in[11];
    const float* w_fc    = (const float*)d_in[12];
    const float* b_fc    = (const float*)d_in[13];
    const float* w_mp    = (const float*)d_in[14];
    const float* b_mp    = (const float*)d_in[15];
    const float* lnf_w   = (const float*)d_in[16];
    const float* lnf_b   = (const float*)d_in[17];

    char* base = (char*)d_ws;
    size_t off = 0;
    auto take = [&](size_t bytes) -> size_t {
        size_t r = off;
        off = (off + bytes + 255) & ~((size_t)255);
        return r;
    };
    float* x    = (float*)(base + take((size_t)cM * cE * 4));
    u16*   xb   = (u16*)  (base + take((size_t)cM * cE * 2));
    u16*   qkvb = (u16*)  (base + take((size_t)cM * 3 * cE * 2));
    u16*   attb = (u16*)  (base + take((size_t)cM * cE * 2));
    u16*   ffb  = (u16*)  (base + take((size_t)cM * cFF * 2));
    u16*   wTb  = (u16*)  (base + take((size_t)cE * cFF * 2));
    u16*   vTb  = (u16*)  (base + take((size_t)cB * cH * cD * cT * 2));
    u16*   wteb = (u16*)  (base + take((size_t)cV * cE * 2));
    float* nllb = (float*)(base + take((size_t)cM * 4));

    float* logits = (float*)d_out;
    float* lossp  = logits + (size_t)cM * cV;

    dim3 b256(256), b128(128);

    embed_kernel<<<(cM * cE) / 256, b256, 0, stream>>>(idx, wte, wpe, x);
    f32_to_bf16_kernel<<<(cV * cE + 255) / 256, b256, 0, stream>>>(wte, wteb, cV * cE);

    for (int l = 0; l < cL; ++l) {
        // ---- attention ----
        ln_kernel<<<cM, b256, 0, stream>>>(x, ln1_w + l * cE, ln1_b + l * cE, xb);
        transp_conv_kernel<<<(cE * 3 * cE + 255) / 256, b256, 0, stream>>>(
            w_qkv + (size_t)l * cE * 3 * cE, wTb, cE, 3 * cE);
        gemm_bf16_kernel<<<dim3((3 * cE) / 64, cM / 256), b256, 0, stream>>>(
            xb, wTb, b_qkv + (size_t)l * 3 * cE, nullptr, qkvb,
            cM, 3 * cE, cE, 3 * cE, 2);
        transpose_v_kernel<<<(cB * cH * cD * cT + 255) / 256, b256, 0, stream>>>(qkvb, vTb);
        attn_kernel<<<cB * cH * 16, b128, 0, stream>>>(qkvb, vTb, attb);
        transp_conv_kernel<<<(cE * cE + 255) / 256, b256, 0, stream>>>(
            w_ap + (size_t)l * cE * cE, wTb, cE, cE);
        gemm_bf16_kernel<<<dim3(cE / 64, cM / 256), b256, 0, stream>>>(
            attb, wTb, b_ap + (size_t)l * cE, x, nullptr,
            cM, cE, cE, cE, 1);
        // ---- MLP ----
        ln_kernel<<<cM, b256, 0, stream>>>(x, ln2_w + l * cE, ln2_b + l * cE, xb);
        transp_conv_kernel<<<(cE * cFF + 255) / 256, b256, 0, stream>>>(
            w_fc + (size_t)l * cE * cFF, wTb, cE, cFF);
        gemm_bf16_kernel<<<dim3(cFF / 64, cM / 256), b256, 0, stream>>>(
            xb, wTb, b_fc + (size_t)l * cFF, nullptr, ffb,
            cM, cFF, cE, cFF, 3);
        transp_conv_kernel<<<(cFF * cE + 255) / 256, b256, 0, stream>>>(
            w_mp + (size_t)l * cFF * cE, wTb, cFF, cE);
        gemm_bf16_kernel<<<dim3(cE / 64, cM / 256), b256, 0, stream>>>(
            ffb, wTb, b_mp + (size_t)l * cE, x, nullptr,
            cM, cE, cFF, cE, 1);
    }

    // ---- final LN + tied lm_head + loss ----
    ln_kernel<<<cM, b256, 0, stream>>>(x, lnf_w, lnf_b, xb);
    gemm_bf16_kernel<<<dim3((cV + 63) / 64, cM / 256), b256, 0, stream>>>(
        xb, wteb, nullptr, logits, nullptr,
        cM, cV, cE, cV, 0);
    loss_row_kernel<<<cM, b256, 0, stream>>>(logits, targets, nllb);
    loss_reduce_kernel<<<1, b256, 0, stream>>>(nllb, targets, lossp);
}
